// MultiReferenceWindowAttention_88854283419798
// MI455X (gfx1250) — matrix operand order repaired and checked by reading, not compiled, not run on hardware
//
#include <hip/hip_runtime.h>
#include <hip/hip_bf16.h>
#include <math.h>

// ---------------------------------------------------------------------------
// MultiReferenceWindowAttention for MI455X (gfx1250, wave32, WMMA).
// All GEMMs use v_wmma_f32_16x16x32_bf16 (bf16 operands, fp32 accumulate).
// ---------------------------------------------------------------------------

typedef __bf16 bf16;
typedef __attribute__((ext_vector_type(16))) __bf16 v16bf;
typedef __attribute__((ext_vector_type(8)))  float  v8f;

#define DEVINL __device__ __forceinline__

DEVINL v8f wmma_bf16(v16bf a, v16bf b, v8f c) {
  // 8 args: (neg_a, A, neg_b, B, c_mod, C, reuse_a, reuse_b)
  return __builtin_amdgcn_wmma_f32_16x16x32_bf16(false, a, false, b,
                                                 (short)0, c, false, false);
}

// A fragment (16x32 bf16) from a row-major [16 x ld] bf16 tile.
// Contiguous k order (same as B frags): lane L -> row M = L&15, K-half = (L>>4)*16;
// VGPR j packs K = khalf + 2j .. +1
DEVINL v16bf a_frag_bf(const bf16* p, int ld) {
  const int L = threadIdx.x & 31;
  const bf16* row = p + (L & 15) * ld + ((L >> 4) << 4);
  v16bf a;
#pragma unroll
  for (int j = 0; j < 8; ++j) {
    const int k0 = 2 * j;
    a[2 * j]     = row[k0];
    a[2 * j + 1] = row[k0 + 1];
  }
  return a;
}

// Same A fragment but sourced from fp32 data (converted on load).
DEVINL v16bf a_frag_f32(const float* p, int ld) {
  const int L = threadIdx.x & 31;
  const float* row = p + (L & 15) * ld + ((L >> 4) << 4);
  v16bf a;
#pragma unroll
  for (int j = 0; j < 8; ++j) {
    const int k0 = 2 * j;
    a[2 * j]     = (bf16)row[k0];
    a[2 * j + 1] = (bf16)row[k0 + 1];
  }
  return a;
}

// B fragment (32x16) where the matrix is stored TRANSPOSED row-major: [16(N) x ld(K)].
// (Used for q*k^T and for y = x @ W^T with W stored (out x in) row-major.)
// Lane L -> col N = L&15, K-half = (L>>4)*16; VGPR j packs K = khalf + 2j, 2j+1.
DEVINL v16bf b_frag_nk(const bf16* p, int ld) {
  const int L = threadIdx.x & 31;
  const bf16* row = p + (L & 15) * ld + ((L >> 4) << 4);
  v16bf b;
#pragma unroll
  for (int j = 0; j < 8; ++j) {
    b[2 * j]     = row[2 * j];
    b[2 * j + 1] = row[2 * j + 1];
  }
  return b;
}

// B fragment (32x16) stored naturally K-major: [32(K) x ld(N)]. (Used for P @ V.)
DEVINL v16bf b_frag_kn(const bf16* p, int ld) {
  const int L = threadIdx.x & 31;
  const bf16* col = p + ((L >> 4) << 4) * ld + (L & 15);
  v16bf b;
#pragma unroll
  for (int j = 0; j < 8; ++j) {
    b[2 * j]     = col[(2 * j) * ld];
    b[2 * j + 1] = col[(2 * j + 1) * ld];
  }
  return b;
}

// ---------------------------------------------------------------------------
// Problem constants
// ---------------------------------------------------------------------------
// C=192, N=64, heads=6, hd=32, B_=128, T=M=4, BT=512, nW=64
#define SCALE 0.17677669529663687f /* 32^-0.5 */

// Workspace layout (bytes). Requires ~290 MB.
static const size_t OFF_WQ  = 0;                      // 192*192 bf16
static const size_t OFF_WKV = OFF_WQ  + 73728;        // 384*192 bf16
static const size_t OFF_WPQ = OFF_WKV + 147456;
static const size_t OFF_WPK = OFF_WPQ + 73728;
static const size_t OFF_WPV = OFF_WPK + 73728;
static const size_t OFF_WPO = OFF_WPV + 73728;
static const size_t OFF_WPR = OFF_WPO + 73728;
static const size_t OFF_QBF = OFF_WPR + 73728;        // (512,6,64,32) bf16
static const size_t OFF_KBF = OFF_QBF + 12582912;     // (512,6,64,32) bf16
static const size_t OFF_VBF = OFF_KBF + 12582912;     // (512,6,64,32) bf16
static const size_t OFF_S1  = OFF_VBF + 12582912;     // 12582912 fp32 (stage1 out, flat b,m,t,h,n,d)
static const size_t OFF_XP  = OFF_S1  + 50331648;     // (512,5,64,192) bf16
static const size_t OFF_Q2  = OFF_XP  + 62914560;     // (512,6,64,32) bf16
static const size_t OFF_K2  = OFF_Q2  + 12582912;     // (512,6,320,32) bf16
static const size_t OFF_V2  = OFF_K2  + 62914560;     // (512,6,320,32) bf16
// S1 is dead after pool-prep; alias pooled-attention outputs into it.
static const size_t OFF_O2  = OFF_S1;                 // (512,64,192) fp32
static const size_t OFF_O3  = OFF_S1 + 25165824;      // (512,64,192) bf16 (post-po, post-gelu)

// ---------------------------------------------------------------------------
// Kernel 0: fp32 -> bf16 weight conversion
// ---------------------------------------------------------------------------
__global__ void k_cvt(const float* __restrict__ s, bf16* __restrict__ d, int n) {
  const int i = blockIdx.x * 256 + threadIdx.x;
  if (i < n) d[i] = (bf16)s[i];
}

// ---------------------------------------------------------------------------
// Kernel 1: q/k/v projections -> head-split bf16 [s][h][n][d]
//   which: 0=q from x, 1=k from x_kv (kv_w rows 0..191), 2=v (rows 192..383)
// ---------------------------------------------------------------------------
__global__ void k_proj_qkv(const float* __restrict__ x, const float* __restrict__ xkv,
                           const bf16* __restrict__ Wq, const bf16* __restrict__ Wkv,
                           const float* __restrict__ qb, const float* __restrict__ kvb,
                           bf16* __restrict__ qo, bf16* __restrict__ ko,
                           bf16* __restrict__ vo) {
  const int s = blockIdx.x;       // 0..511
  const int which = blockIdx.y;   // 0,1,2
  const float* src = (which == 0 ? x : xkv) + (size_t)s * 64 * 192;
  const bf16* W = (which == 0) ? Wq : (which == 1 ? Wkv : Wkv + 192 * 192);
  const float* bias = (which == 0) ? qb : (which == 1 ? kvb : kvb + 192);
  bf16* out = ((which == 0) ? qo : (which == 1 ? ko : vo)) + (size_t)s * 6 * 64 * 32;

  const int wave = threadIdx.x >> 5;
  const int L = threadIdx.x & 31;
  const int hi = (L >> 4) << 3;
  const int nn = L & 15;
#pragma unroll
  for (int i = 0; i < 6; ++i) {
    const int tt = wave * 6 + i;        // 48 tiles: 4 row x 12 col
    const int rt = tt / 12, ct = tt % 12;
    v8f acc = {};
#pragma unroll
    for (int kt = 0; kt < 6; ++kt) {    // K = 192
      v16bf a = a_frag_f32(src + rt * 16 * 192 + kt * 32, 192);
      v16bf b = b_frag_nk(W + ct * 16 * 192 + kt * 32, 192);
      acc = wmma_bf16(a, b, acc);
    }
    const int col = ct * 16 + nn;       // output channel
    const int h = col >> 5, d = col & 31;
    const float bb = bias[col];
#pragma unroll
    for (int r = 0; r < 8; ++r) {
      const int n = rt * 16 + r + hi;
      out[(h * 64 + n) * 32 + d] = (bf16)(acc[r] + bb);
    }
  }
}

// ---------------------------------------------------------------------------
// Kernel 2: stage-1 window attention. One block per (b_,t,m,h); 4 waves,
// each wave owns a 16-row band of the 64x64 score matrix.
// ---------------------------------------------------------------------------
__global__ void k_attn1(const bf16* __restrict__ qbf, const bf16* __restrict__ kbf,
                        const bf16* __restrict__ vbf, const float* __restrict__ rpb,
                        const float* __restrict__ mask, float* __restrict__ s1) {
  int id = blockIdx.x;
  const int h = id % 6; id /= 6;
  const int m = id % 4; id /= 4;
  const int t = id % 4; id /= 4;
  const int b_ = id;                    // 0..127
  const int bt = b_ * 4 + t;
  const int bm = b_ * 4 + m;
  const bf16* q = qbf + (size_t)(bt * 6 + h) * 2048;
  const bf16* k = kbf + (size_t)(bm * 6 + h) * 2048;
  const bf16* v = vbf + (size_t)(bm * 6 + h) * 2048;

  const int wave = threadIdx.x >> 5;
  const int m16 = wave * 16;
  const int L = threadIdx.x & 31;
  const int hi = (L >> 4) << 3;
  const int nn = L & 15;

  // scores: 4 tiles of 16x16, single wmma each (K = hd = 32)
  v16bf aq = a_frag_bf(q + m16 * 32, 32);
  float sc[4][8];
#pragma unroll
  for (int nt = 0; nt < 4; ++nt) {
    v16bf bk = b_frag_nk(k + nt * 16 * 32, 32);
    v8f c = {};
    c = wmma_bf16(aq, bk, c);
#pragma unroll
    for (int r = 0; r < 8; ++r) sc[nt][r] = c[r];
  }

  // fp32 epilogue: scale + relative-position bias + shifted-window mask
  const int w = b_ & 63;
#pragma unroll
  for (int nt = 0; nt < 4; ++nt)
#pragma unroll
    for (int r = 0; r < 8; ++r) {
      const int n = m16 + r + hi;       // query token
      const int e = nt * 16 + nn;       // key token
      const int rpi = ((n >> 3) - (e >> 3) + 7) * 15 + ((n & 7) - (e & 7) + 7);
      sc[nt][r] = sc[nt][r] * SCALE + rpb[rpi * 6 + h] + mask[(w * 64 + n) * 64 + e];
    }

  // softmax over the 64 keys: 4 register tiles + xor-shuffle over 16 lanes/half
#pragma unroll
  for (int r = 0; r < 8; ++r) {
    float mx = fmaxf(fmaxf(sc[0][r], sc[1][r]), fmaxf(sc[2][r], sc[3][r]));
#pragma unroll
    for (int off = 1; off < 16; off <<= 1) mx = fmaxf(mx, __shfl_xor(mx, off, 32));
    float sum = 0.f;
#pragma unroll
    for (int nt = 0; nt < 4; ++nt) { sc[nt][r] = __expf(sc[nt][r] - mx); sum += sc[nt][r]; }
#pragma unroll
    for (int off = 1; off < 16; off <<= 1) sum += __shfl_xor(sum, off, 32);
    const float inv = 1.0f / sum;
#pragma unroll
    for (int nt = 0; nt < 4; ++nt) sc[nt][r] *= inv;
  }

  // re-stage P into A-fragment-friendly LDS layout (wave-private rows, no barrier)
  __shared__ bf16 P[64 * 72];
#pragma unroll
  for (int nt = 0; nt < 4; ++nt)
#pragma unroll
    for (int r = 0; r < 8; ++r)
      P[(m16 + r + hi) * 72 + nt * 16 + nn] = (bf16)sc[nt][r];

  // out = P(16x64) x V(64x32); store fp32 in flat (b_, m, t, h, n, d) order
  const size_t obase = (((size_t)(b_ * 4 + m) * 4 + t) * 6 + h) * 2048;
#pragma unroll
  for (int dt = 0; dt < 2; ++dt) {
    v8f c = {};
#pragma unroll
    for (int kt = 0; kt < 2; ++kt) {
      v16bf a = a_frag_bf(&P[m16 * 72 + kt * 32], 72);
      v16bf b = b_frag_kn(v + kt * 32 * 32 + dt * 16, 32);
      c = wmma_bf16(a, b, c);
    }
#pragma unroll
    for (int r = 0; r < 8; ++r) {
      const int n = m16 + r + hi;
      s1[obase + n * 32 + dt * 16 + nn] = c[r];
    }
  }
}

// ---------------------------------------------------------------------------
// Kernel 3: pooled-sequence prep: mean over M, +pos_emb, convert to bf16.
// s1 viewed flat as (BT=512, M=4, 12288); xp is (512, 5, 12288).
// ---------------------------------------------------------------------------
__global__ void k_poolprep(const float* __restrict__ s1, const float* __restrict__ pe,
                           bf16* __restrict__ xp) {
  const int bt = blockIdx.x;
  const size_t ib = (size_t)bt * 4 * 12288;
  const size_t ob = (size_t)bt * 5 * 12288;
  for (int i = threadIdx.x; i < 12288; i += 256) {
    const float p = pe[i];
    const float v0 = s1[ib + i];
    const float v1 = s1[ib + 12288 + i];
    const float v2 = s1[ib + 24576 + i];
    const float v3 = s1[ib + 36864 + i];
    xp[ob + i]         = (bf16)((v0 + v1 + v2 + v3) * 0.25f + p);
    xp[ob + 12288 + i] = (bf16)(v0 + p);
    xp[ob + 24576 + i] = (bf16)(v1 + p);
    xp[ob + 36864 + i] = (bf16)(v2 + p);
    xp[ob + 49152 + i] = (bf16)(v3 + p);
  }
}

// ---------------------------------------------------------------------------
// Kernel 4: pooled q/k/v projections. which: 0=q2 (64 rows), 1=k2, 2=v2 (320 rows)
// Output head-split: [bt][h][token][d]
// ---------------------------------------------------------------------------
__global__ void k_proj_pool(const bf16* __restrict__ xp,
                            const bf16* __restrict__ Wpq, const bf16* __restrict__ Wpk,
                            const bf16* __restrict__ Wpv,
                            const float* __restrict__ bpq, const float* __restrict__ bpk,
                            const float* __restrict__ bpv,
                            bf16* __restrict__ q2, bf16* __restrict__ k2,
                            bf16* __restrict__ v2) {
  const int bt = blockIdx.x;
  const int which = blockIdx.y;
  const bf16* src = xp + (size_t)bt * 5 * 12288;   // 320 x 192 (q2 uses first 64 rows)
  const bf16* W;
  const float* bias;
  bf16* out;
  int rows;
  if (which == 0)      { W = Wpq; bias = bpq; rows = 64;  out = q2 + (size_t)bt * 6 * 64 * 32; }
  else if (which == 1) { W = Wpk; bias = bpk; rows = 320; out = k2 + (size_t)bt * 6 * 320 * 32; }
  else                 { W = Wpv; bias = bpv; rows = 320; out = v2 + (size_t)bt * 6 * 320 * 32; }

  const int total = (rows / 16) * 12;
  const int wave = threadIdx.x >> 5;
  const int L = threadIdx.x & 31;
  const int hi = (L >> 4) << 3, nn = L & 15;
  for (int tt = wave; tt < total; tt += 8) {
    const int rt = tt / 12, ct = tt % 12;
    v8f acc = {};
#pragma unroll
    for (int kt = 0; kt < 6; ++kt) {
      v16bf a = a_frag_bf(src + rt * 16 * 192 + kt * 32, 192);
      v16bf b = b_frag_nk(W + ct * 16 * 192 + kt * 32, 192);
      acc = wmma_bf16(a, b, acc);
    }
    const int col = ct * 16 + nn;
    const int h = col >> 5, d = col & 31;
    const float bb = bias[col];
#pragma unroll
    for (int r = 0; r < 8; ++r) {
      const int n = rt * 16 + r + hi;
      out[((size_t)h * rows + n) * 32 + d] = (bf16)(acc[r] + bb);
    }
  }
}

// ---------------------------------------------------------------------------
// Kernel 5: pooled attention: per (bt,h), 64 queries x 320 keys, hd=32.
// ---------------------------------------------------------------------------
__global__ void k_attn2(const bf16* __restrict__ q2, const bf16* __restrict__ k2,
                        const bf16* __restrict__ v2, float* __restrict__ o2) {
  const int bt = blockIdx.x / 6;
  const int h = blockIdx.x % 6;
  const bf16* q = q2 + ((size_t)bt * 6 + h) * 2048;
  const bf16* k = k2 + ((size_t)bt * 6 + h) * 10240;
  const bf16* v = v2 + ((size_t)bt * 6 + h) * 10240;
  const int wave = threadIdx.x >> 5;
  const int m16 = wave * 16;
  const int L = threadIdx.x & 31;
  const int hi = (L >> 4) << 3, nn = L & 15;

  v16bf aq = a_frag_bf(q + m16 * 32, 32);
  float sc[20][8];
#pragma unroll
  for (int nt = 0; nt < 20; ++nt) {
    v16bf bk = b_frag_nk(k + nt * 16 * 32, 32);
    v8f c = {};
    c = wmma_bf16(aq, bk, c);
#pragma unroll
    for (int r = 0; r < 8; ++r) sc[nt][r] = c[r] * SCALE;
  }
#pragma unroll
  for (int r = 0; r < 8; ++r) {
    float mx = -1e30f;
#pragma unroll
    for (int nt = 0; nt < 20; ++nt) mx = fmaxf(mx, sc[nt][r]);
#pragma unroll
    for (int off = 1; off < 16; off <<= 1) mx = fmaxf(mx, __shfl_xor(mx, off, 32));
    float sum = 0.f;
#pragma unroll
    for (int nt = 0; nt < 20; ++nt) { sc[nt][r] = __expf(sc[nt][r] - mx); sum += sc[nt][r]; }
#pragma unroll
    for (int off = 1; off < 16; off <<= 1) sum += __shfl_xor(sum, off, 32);
    const float inv = 1.0f / sum;
#pragma unroll
    for (int nt = 0; nt < 20; ++nt) sc[nt][r] *= inv;
  }

  __shared__ bf16 P[64 * 328];
#pragma unroll
  for (int nt = 0; nt < 20; ++nt)
#pragma unroll
    for (int r = 0; r < 8; ++r)
      P[(m16 + r + hi) * 328 + nt * 16 + nn] = (bf16)sc[nt][r];

#pragma unroll
  for (int dt = 0; dt < 2; ++dt) {
    v8f c = {};
#pragma unroll
    for (int kt = 0; kt < 10; ++kt) {   // K = 320
      v16bf a = a_frag_bf(&P[m16 * 328 + kt * 32], 328);
      v16bf b = b_frag_kn(v + kt * 32 * 32 + dt * 16, 32);
      c = wmma_bf16(a, b, c);
    }
#pragma unroll
    for (int r = 0; r < 8; ++r) {
      const int n = m16 + r + hi;
      o2[((size_t)bt * 64 + n) * 192 + h * 32 + dt * 16 + nn] = c[r];
    }
  }
}

// ---------------------------------------------------------------------------
// Kernel 6: o3 = bf16(gelu(o2 @ po_w^T + po_b))
// ---------------------------------------------------------------------------
__global__ void k_po_gelu(const float* __restrict__ o2, const bf16* __restrict__ Wpo,
                          const float* __restrict__ bpo, bf16* __restrict__ o3) {
  const int bt = blockIdx.x;
  const float* src = o2 + (size_t)bt * 12288;
  bf16* out = o3 + (size_t)bt * 12288;
  const int wave = threadIdx.x >> 5;
  const int L = threadIdx.x & 31;
  const int hi = (L >> 4) << 3, nn = L & 15;
#pragma unroll
  for (int i = 0; i < 6; ++i) {
    const int tt = wave * 6 + i;
    const int rt = tt / 12, ct = tt % 12;
    v8f acc = {};
#pragma unroll
    for (int kt = 0; kt < 6; ++kt) {
      v16bf a = a_frag_f32(src + rt * 16 * 192 + kt * 32, 192);
      v16bf b = b_frag_nk(Wpo + ct * 16 * 192 + kt * 32, 192);
      acc = wmma_bf16(a, b, acc);
    }
    const int col = ct * 16 + nn;
    const float bb = bpo[col];
#pragma unroll
    for (int r = 0; r < 8; ++r) {
      const int n = rt * 16 + r + hi;
      float y = acc[r] + bb;
      y = 0.5f * y * (1.0f + erff(y * 0.70710678118654752f));  // exact gelu
      out[n * 192 + col] = (bf16)y;
    }
  }
}

// ---------------------------------------------------------------------------
// Kernel 7: out = gelu_result @ proj_w^T + proj_b  (fp32 output)
// ---------------------------------------------------------------------------
__global__ void k_final(const bf16* __restrict__ g, const bf16* __restrict__ Wpr,
                        const float* __restrict__ bpr, float* __restrict__ out) {
  const int bt = blockIdx.x;
  const bf16* src = g + (size_t)bt * 12288;
  float* dst = out + (size_t)bt * 12288;
  const int wave = threadIdx.x >> 5;
  const int L = threadIdx.x & 31;
  const int hi = (L >> 4) << 3, nn = L & 15;
#pragma unroll
  for (int i = 0; i < 6; ++i) {
    const int tt = wave * 6 + i;
    const int rt = tt / 12, ct = tt % 12;
    v8f acc = {};
#pragma unroll
    for (int kt = 0; kt < 6; ++kt) {
      v16bf a = a_frag_bf(src + rt * 16 * 192 + kt * 32, 192);
      v16bf b = b_frag_nk(Wpr + ct * 16 * 192 + kt * 32, 192);
      acc = wmma_bf16(a, b, acc);
    }
    const int col = ct * 16 + nn;
    const float bb = bpr[col];
#pragma unroll
    for (int r = 0; r < 8; ++r) {
      const int n = rt * 16 + r + hi;
      dst[n * 192 + col] = acc[r] + bb;
    }
  }
}

// ---------------------------------------------------------------------------
extern "C" void kernel_launch(void* const* d_in, const int* in_sizes, int n_in,
                              void* d_out, int out_size, void* d_ws, size_t ws_size,
                              hipStream_t stream) {
  const float* x    = (const float*)d_in[0];
  const float* xkv  = (const float*)d_in[1];
  const float* mask = (const float*)d_in[2];
  const float* rpb  = (const float*)d_in[3];
  const float* q_w  = (const float*)d_in[4];
  const float* q_b  = (const float*)d_in[5];
  const float* kv_w = (const float*)d_in[6];
  const float* kv_b = (const float*)d_in[7];
  const float* pe   = (const float*)d_in[8];
  const float* pq_w = (const float*)d_in[9];
  const float* pq_b = (const float*)d_in[10];
  const float* pk_w = (const float*)d_in[11];
  const float* pk_b = (const float*)d_in[12];
  const float* pv_w = (const float*)d_in[13];
  const float* pv_b = (const float*)d_in[14];
  const float* po_w = (const float*)d_in[15];
  const float* po_b = (const float*)d_in[16];
  const float* pr_w = (const float*)d_in[17];
  const float* pr_b = (const float*)d_in[18];

  char* ws = (char*)d_ws;
  bf16* WQ  = (bf16*)(ws + OFF_WQ);
  bf16* WKV = (bf16*)(ws + OFF_WKV);
  bf16* WPQ = (bf16*)(ws + OFF_WPQ);
  bf16* WPK = (bf16*)(ws + OFF_WPK);
  bf16* WPV = (bf16*)(ws + OFF_WPV);
  bf16* WPO = (bf16*)(ws + OFF_WPO);
  bf16* WPR = (bf16*)(ws + OFF_WPR);
  bf16* QBF = (bf16*)(ws + OFF_QBF);
  bf16* KBF = (bf16*)(ws + OFF_KBF);
  bf16* VBF = (bf16*)(ws + OFF_VBF);
  float* S1 = (float*)(ws + OFF_S1);
  bf16* XP  = (bf16*)(ws + OFF_XP);
  bf16* Q2  = (bf16*)(ws + OFF_Q2);
  bf16* K2  = (bf16*)(ws + OFF_K2);
  bf16* V2  = (bf16*)(ws + OFF_V2);
  float* O2 = (float*)(ws + OFF_O2);
  bf16* O3  = (bf16*)(ws + OFF_O3);

  // 0) weight conversion fp32 -> bf16
  k_cvt<<<(36864 + 255) / 256, 256, 0, stream>>>(q_w,  WQ,  36864);
  k_cvt<<<(73728 + 255) / 256, 256, 0, stream>>>(kv_w, WKV, 73728);
  k_cvt<<<(36864 + 255) / 256, 256, 0, stream>>>(pq_w, WPQ, 36864);
  k_cvt<<<(36864 + 255) / 256, 256, 0, stream>>>(pk_w, WPK, 36864);
  k_cvt<<<(36864 + 255) / 256, 256, 0, stream>>>(pv_w, WPV, 36864);
  k_cvt<<<(36864 + 255) / 256, 256, 0, stream>>>(po_w, WPO, 36864);
  k_cvt<<<(36864 + 255) / 256, 256, 0, stream>>>(pr_w, WPR, 36864);

  // 1) q/k/v projections (WMMA)
  k_proj_qkv<<<dim3(512, 3), 256, 0, stream>>>(x, xkv, WQ, WKV, q_b, kv_b,
                                               QBF, KBF, VBF);
  // 2) stage-1 windowed attention (WMMA x2 stages + softmax)
  k_attn1<<<12288, 128, 0, stream>>>(QBF, KBF, VBF, rpb, mask, S1);
  // 3) pooling prep: mean over M, +pos_emb, bf16
  k_poolprep<<<512, 256, 0, stream>>>(S1, pe, XP);
  // 4) pooled q/k/v projections (WMMA)
  k_proj_pool<<<dim3(512, 3), 256, 0, stream>>>(XP, WPQ, WPK, WPV,
                                                pq_b, pk_b, pv_b, Q2, K2, V2);
  // 5) pooled attention 64x320 (WMMA + softmax)
  k_attn2<<<3072, 128, 0, stream>>>(Q2, K2, V2, O2);
  // 6) po projection + exact gelu (WMMA)
  k_po_gelu<<<512, 256, 0, stream>>>(O2, WPO, po_b, O3);
  // 7) final projection -> d_out fp32 (WMMA)
  k_final<<<512, 256, 0, stream>>>(O3, WPR, pr_b, (float*)d_out);
}
